// HashSAT_35862976921619
// MI455X (gfx1250) — matrix operand order repaired and checked by reading, not compiled, not run on hardware
//
#include <hip/hip_runtime.h>
#include <cstddef>

typedef __bf16 v16bf __attribute__((ext_vector_type(16)));
typedef float  v8f   __attribute__((ext_vector_type(8)));

#define H 128
#define NCOLOR 3

// ---------------------------------------------------------------------------
// In/out degree histograms (int atomics, one-time cost)
// ---------------------------------------------------------------------------
__global__ void hist_kernel(const int* __restrict__ src, const int* __restrict__ dst,
                            int* __restrict__ cnt_out, int* __restrict__ cnt_in, int E) {
    int e = blockIdx.x * blockDim.x + threadIdx.x;
    if (e >= E) return;
    atomicAdd(&cnt_out[src[e]], 1);
    atomicAdd(&cnt_in[dst[e]], 1);
}

// n[i] = rsqrt(max(cnt[i],1))  over contiguous [cnt_out|cnt_in] -> [n_out|n_in]
__global__ void deg2norm_kernel(const int* __restrict__ cnt, float* __restrict__ n, int total) {
    int i = blockIdx.x * blockDim.x + threadIdx.x;
    if (i >= total) return;
    n[i] = rsqrtf(fmaxf((float)cnt[i], 1.0f));
}

// ---------------------------------------------------------------------------
// Two-level exclusive scan of in-degree counts -> CSR row offsets
// ---------------------------------------------------------------------------
__global__ void scan1_kernel(const int* __restrict__ cnt, int* __restrict__ excl,
                             int* __restrict__ bsum, int N) {
    __shared__ int sh[256];
    int i = blockIdx.x * 256 + threadIdx.x;
    int v = (i < N) ? cnt[i] : 0;
    sh[threadIdx.x] = v;
    __syncthreads();
#pragma unroll
    for (int off = 1; off < 256; off <<= 1) {
        int t = (threadIdx.x >= off) ? sh[threadIdx.x - off] : 0;
        __syncthreads();
        sh[threadIdx.x] += t;
        __syncthreads();
    }
    if (i < N) excl[i] = sh[threadIdx.x] - v;
    if (threadIdx.x == 255) bsum[blockIdx.x] = sh[255];
}

__global__ void scan2_kernel(int* __restrict__ bsum, int nb) {
    __shared__ int sh[512];
    int v = (threadIdx.x < nb) ? bsum[threadIdx.x] : 0;
    sh[threadIdx.x] = v;
    __syncthreads();
#pragma unroll
    for (int off = 1; off < 512; off <<= 1) {
        int t = (threadIdx.x >= off) ? sh[threadIdx.x - off] : 0;
        __syncthreads();
        sh[threadIdx.x] += t;
        __syncthreads();
    }
    if (threadIdx.x < nb) bsum[threadIdx.x] = sh[threadIdx.x] - v;  // exclusive
}

__global__ void scan3_kernel(int* __restrict__ excl, const int* __restrict__ bsum,
                             int N, int E) {
    int i = blockIdx.x * blockDim.x + threadIdx.x;
    if (i < N) excl[i] += bsum[i >> 8];
    if (i == 0) excl[N] = E;
}

__global__ void copy_kernel(const int* __restrict__ a, int* __restrict__ b, int n) {
    int i = blockIdx.x * blockDim.x + threadIdx.x;
    if (i < n) b[i] = a[i];
}

// Bucket fill: edge source ids grouped by destination node
__global__ void fill_kernel(const int* __restrict__ src, const int* __restrict__ dst,
                            int* __restrict__ next, int* __restrict__ esrc, int E) {
    int e = blockIdx.x * blockDim.x + threadIdx.x;
    if (e >= E) return;
    int pos = atomicAdd(&next[dst[e]], 1);
    esrc[pos] = src[e];
}

// ---------------------------------------------------------------------------
// Pack fp32 weights [L1][128][128] into bf16 WMMA B-operand layout:
// wpack[l][ntile t][kstep s][lane][e] = W[l][s*32 + (lane>>4)*16 + e][t*16 + (lane&15)]
// ---------------------------------------------------------------------------
__global__ void pack_w_kernel(const float* __restrict__ Ws, __bf16* __restrict__ wpack, int total) {
    int t = blockIdx.x * blockDim.x + threadIdx.x;
    if (t >= total) return;
    int l    = t / 16384;
    int r    = t % 16384;
    int nt   = r / 2048;
    int s    = (r % 2048) / 512;
    int lane = (r % 512) / 16;
    int e    = r % 16;
    int k    = s * 32 + ((lane >> 4) * 16) + e;
    int n    = nt * 16 + (lane & 15);
    wpack[t] = (__bf16)Ws[(size_t)l * 16384 + (size_t)k * H + n];
}

// ---------------------------------------------------------------------------
// GEMM: hw[N,128] = (h * n_out[:,None]) @ W  via v_wmma_f32_16x16x32_bf16.
// B (32KB packed weights) staged in LDS once per block (8 waves share it);
// each wave owns one 16-row tile and all 8 column tiles.
// ---------------------------------------------------------------------------
__global__ void gemm_wmma_kernel(const float* __restrict__ h,
                                 const float* __restrict__ n_out,
                                 const __bf16* __restrict__ wpackL,
                                 float* __restrict__ hw, int ntiles) {
    __shared__ __bf16 bsh[16384];  // 32 KB (WGP has 320 KB LDS)

    // cooperative stage of packed B into LDS (256 threads x 8 uint4)
    {
        const uint4* gsrc = (const uint4*)wpackL;
        uint4* ldst = (uint4*)bsh;
#pragma unroll
        for (int i = 0; i < 8; ++i)
            ldst[threadIdx.x + 256 * i] = gsrc[threadIdx.x + 256 * i];
    }
    __syncthreads();

    int wave = threadIdx.x >> 5;
    int lane = threadIdx.x & 31;
    int tile = blockIdx.x * 8 + wave;
    if (tile >= ntiles) return;

    int rowA  = tile * 16 + (lane & 15);
    int offK  = (lane >> 4) * 16;
    float scl = n_out[rowA];

    v8f acc0 = {}, acc1 = {}, acc2 = {}, acc3 = {};
    v8f acc4 = {}, acc5 = {}, acc6 = {}, acc7 = {};

#pragma unroll
    for (int s = 0; s < 4; ++s) {
        // load + scale + convert A (16x32 bf16, contiguous k = 16*(lane>>4) + e, same as packed B)
        const float* hp = h + (size_t)rowA * H + s * 32;
        v16bf a;
#pragma unroll
        for (int p = 0; p < 8; ++p) {
            int k = offK + 2 * p;
            float2 v = *(const float2*)(hp + k);
            a[2 * p]     = (__bf16)(v.x * scl);
            a[2 * p + 1] = (__bf16)(v.y * scl);
        }
        const __bf16* bbase = bsh + s * 512 + lane * 16;
#pragma unroll
        for (int t = 0; t < 8; ++t) {
            v16bf b = *(const v16bf*)(bbase + t * 2048);
            v8f* accp = (t == 0) ? &acc0 : (t == 1) ? &acc1 : (t == 2) ? &acc2 :
                        (t == 3) ? &acc3 : (t == 4) ? &acc4 : (t == 5) ? &acc5 :
                        (t == 6) ? &acc6 : &acc7;
            *accp = __builtin_amdgcn_wmma_f32_16x16x32_bf16(
                false, a, false, b, (short)0, *accp, false, false);
        }
    }

    int n0 = lane & 15;
    int m0 = (lane >> 4) * 8;
    float* outbase = hw + (size_t)(tile * 16 + m0) * H + n0;
#pragma unroll
    for (int r = 0; r < 8; ++r) {
        float* row = outbase + (size_t)r * H;
        row[0 * 16] = acc0[r]; row[1 * 16] = acc1[r];
        row[2 * 16] = acc2[r]; row[3 * 16] = acc3[r];
        row[4 * 16] = acc4[r]; row[5 * 16] = acc5[r];
        row[6 * 16] = acc6[r]; row[7 * 16] = acc7[r];
    }
}

// ---------------------------------------------------------------------------
// Pull-mode aggregation (atomic-free): one wave per dst node, lanes split the
// 128 features; sweep CSR in-edge list with coalesced 512B row reads, fused
// with n_in scale + bias + relu.
// ---------------------------------------------------------------------------
__global__ void gather_bias_relu_kernel(const float* __restrict__ hw,
                                        const int* __restrict__ esrc,
                                        const int* __restrict__ row_off,
                                        const float* __restrict__ n_in,
                                        const float* __restrict__ b,
                                        float* __restrict__ h, int N) {
    int t = blockIdx.x * blockDim.x + threadIdx.x;
    int i = t >> 5;
    if (i >= N) return;
    int j = (t & 31) * 4;
    int p0 = row_off[i], p1 = row_off[i + 1];
    float4 acc = make_float4(0.f, 0.f, 0.f, 0.f);
    for (int p = p0; p < p1; ++p) {
        int s = esrc[p];  // broadcast across the wave
        float4 v = *(const float4*)(hw + (size_t)s * H + j);
        acc.x += v.x; acc.y += v.y; acc.z += v.z; acc.w += v.w;
    }
    float ni = n_in[i];
    float4 bb = *(const float4*)(b + j);
    float4 o;
    o.x = fmaxf(fmaf(acc.x, ni, bb.x), 0.f);
    o.y = fmaxf(fmaf(acc.y, ni, bb.y), 0.f);
    o.z = fmaxf(fmaf(acc.z, ni, bb.z), 0.f);
    o.w = fmaxf(fmaf(acc.w, ni, bb.w), 0.f);
    *(float4*)(h + (size_t)i * H + j) = o;
}

// Final layer GEMV: hw3[i,c] = sum_k h[i,k]*n_out[i]*W_last[k,c]
__global__ void gemv_last_kernel(const float* __restrict__ h, const float* __restrict__ n_out,
                                 const float* __restrict__ Wl, float* __restrict__ hw3, int N) {
    int i = blockIdx.x * blockDim.x + threadIdx.x;
    if (i >= N) return;
    float s = n_out[i];
    const float* hp = h + (size_t)i * H;
    float a0 = 0.f, a1 = 0.f, a2 = 0.f;
#pragma unroll 4
    for (int k = 0; k < H; ++k) {
        float v = hp[k] * s;
        a0 = fmaf(v, Wl[k * 3 + 0], a0);
        a1 = fmaf(v, Wl[k * 3 + 1], a1);
        a2 = fmaf(v, Wl[k * 3 + 2], a2);
    }
    hw3[(size_t)i * 3 + 0] = a0;
    hw3[(size_t)i * 3 + 1] = a1;
    hw3[(size_t)i * 3 + 2] = a2;
}

// ---------------------------------------------------------------------------
// Final layer pull-gather + per-node softmax -> colors, fused with
// attention-pool partials (wave32 shuffle reduce, 4 atomics per wave).
// gate is bounded (colors in [0,1], small pool_w) -> exp w/o shift is safe.
// ---------------------------------------------------------------------------
__global__ void last_softmax_pool_kernel(const float* __restrict__ hw3,
                                         const int* __restrict__ esrc,
                                         const int* __restrict__ row_off,
                                         const float* __restrict__ n_in,
                                         const float* __restrict__ b_last,
                                         const float* __restrict__ pool_w,
                                         const float* __restrict__ pool_b,
                                         float* __restrict__ colors,
                                         float* __restrict__ stats, int N) {
    int i = blockIdx.x * blockDim.x + threadIdx.x;
    bool valid = (i < N);
    float g = 0.f, gc0 = 0.f, gc1 = 0.f, gc2 = 0.f;
    if (valid) {
        float a0 = 0.f, a1 = 0.f, a2 = 0.f;
        int p0 = row_off[i], p1 = row_off[i + 1];
        for (int p = p0; p < p1; ++p) {
            int s = esrc[p];
            a0 += hw3[(size_t)s * 3 + 0];
            a1 += hw3[(size_t)s * 3 + 1];
            a2 += hw3[(size_t)s * 3 + 2];
        }
        float ni = n_in[i];
        float l0 = fmaf(a0, ni, b_last[0]);
        float l1 = fmaf(a1, ni, b_last[1]);
        float l2 = fmaf(a2, ni, b_last[2]);
        float mx = fmaxf(l0, fmaxf(l1, l2));
        float e0 = __expf(l0 - mx), e1 = __expf(l1 - mx), e2 = __expf(l2 - mx);
        float inv = 1.0f / (e0 + e1 + e2);
        float c0 = e0 * inv, c1 = e1 * inv, c2 = e2 * inv;
        colors[(size_t)i * 3 + 0] = c0;
        colors[(size_t)i * 3 + 1] = c1;
        colors[(size_t)i * 3 + 2] = c2;
        float gate = c0 * pool_w[0] + c1 * pool_w[1] + c2 * pool_w[2] + pool_b[0];
        g = __expf(gate);
        gc0 = g * c0; gc1 = g * c1; gc2 = g * c2;
    }
#pragma unroll
    for (int o = 16; o > 0; o >>= 1) {
        g   += __shfl_down(g, o);
        gc0 += __shfl_down(gc0, o);
        gc1 += __shfl_down(gc1, o);
        gc2 += __shfl_down(gc2, o);
    }
    if ((threadIdx.x & 31) == 0) {
        atomicAdd(&stats[0], g);
        atomicAdd(&stats[1], gc0);
        atomicAdd(&stats[2], gc1);
        atomicAdd(&stats[3], gc2);
    }
}

__global__ void finalize_kernel(const float* __restrict__ stats, const float* __restrict__ out_w,
                                const float* __restrict__ out_b, float* __restrict__ sat) {
    float invZ = 1.0f / stats[0];
    float v = stats[1] * invZ * out_w[0] + stats[2] * invZ * out_w[1] +
              stats[3] * invZ * out_w[2] + out_b[0];
    sat[0] = 1.0f / (1.0f + __expf(-v));
}

// ---------------------------------------------------------------------------
extern "C" void kernel_launch(void* const* d_in, const int* in_sizes, int n_in_cnt,
                              void* d_out, int out_size, void* d_ws, size_t ws_size,
                              hipStream_t stream) {
    const float* x      = (const float*)d_in[0];
    const int*   src    = (const int*)d_in[1];
    const int*   dst    = (const int*)d_in[2];
    const float* Ws     = (const float*)d_in[3];
    const float* bs     = (const float*)d_in[4];
    const float* W_last = (const float*)d_in[5];
    const float* b_last = (const float*)d_in[6];
    const float* pool_w = (const float*)d_in[7];
    const float* pool_b = (const float*)d_in[8];
    const float* out_w  = (const float*)d_in[9];
    const float* out_b  = (const float*)d_in[10];

    const int N  = in_sizes[0] / H;        // 100000
    const int E  = in_sizes[1];            // 1600000
    const int L1 = in_sizes[3] / (H * H);  // 7
    const int ntiles = N / 16;             // N % 16 == 0
    const int NB = (N + 255) / 256;        // scan blocks (<= 512)

    // ---- workspace layout ----
    char* wsb = (char*)d_ws;
    __bf16* wpack = (__bf16*)wsb;  // L1*16384 bf16, 32B-aligned chunks
    size_t off = ((size_t)L1 * 16384 * sizeof(__bf16) + 255) & ~(size_t)255;

    int* cnt     = (int*)(wsb + off);      // 2N: [cnt_out | cnt_in]
    int* row_off = cnt + (size_t)2 * N;    // N+1
    int* bsum    = row_off + N + 1;        // 512
    int* next    = bsum + 512;             // N
    int* esrc    = next + N;               // E
    size_t ioff = off + ((size_t)(2 * N + N + 1 + 512 + N + E)) * sizeof(int);
    ioff = (ioff + 255) & ~(size_t)255;

    float* n_out = (float*)(wsb + ioff);        // N
    float* n_in_ = n_out + N;                   // N (contiguous)
    float* h     = n_in_ + N;                   // N*128
    float* hw    = h + (size_t)N * H;           // N*128
    float* hw3   = hw + (size_t)N * H;          // N*3
    float* stats = hw3 + (size_t)N * 3;         // 4 (+pad)

    float* colors = (float*)d_out;              // N*3
    float* sat    = colors + (size_t)N * 3;     // 1

    const int TB = 256;

    // ---- degrees, norms, CSR-by-dst build (one-time) ----
    hipMemsetAsync(cnt, 0, (size_t)2 * N * sizeof(int), stream);
    hipMemsetAsync(stats, 0, 4 * sizeof(float), stream);
    hist_kernel<<<(E + TB - 1) / TB, TB, 0, stream>>>(src, dst, cnt, cnt + N, E);
    deg2norm_kernel<<<(2 * N + TB - 1) / TB, TB, 0, stream>>>(cnt, n_out, 2 * N);

    scan1_kernel<<<NB, 256, 0, stream>>>(cnt + N, row_off, bsum, N);
    scan2_kernel<<<1, 512, 0, stream>>>(bsum, NB);
    scan3_kernel<<<(N + TB) / TB, TB, 0, stream>>>(row_off, bsum, N, E);
    copy_kernel<<<(N + TB - 1) / TB, TB, 0, stream>>>(row_off, next, N);
    fill_kernel<<<(E + TB - 1) / TB, TB, 0, stream>>>(src, dst, next, esrc, E);

    // ---- pack weights to bf16 WMMA layout ----
    int wtot = L1 * 16384;
    pack_w_kernel<<<(wtot + TB - 1) / TB, TB, 0, stream>>>(Ws, wpack, wtot);

    // ---- hidden layers: WMMA GEMM + atomic-free pull aggregation ----
    for (int l = 0; l < L1; ++l) {
        const float* hsrc = (l == 0) ? x : h;
        gemm_wmma_kernel<<<(ntiles + 7) / 8, TB, 0, stream>>>(
            hsrc, n_out, wpack + (size_t)l * 16384, hw, ntiles);
        gather_bias_relu_kernel<<<(N * 32 + TB - 1) / TB, TB, 0, stream>>>(
            hw, esrc, row_off, n_in_, bs + (size_t)l * H, h, N);
    }

    // ---- final layer (128 -> 3) + fused softmax/pool ----
    gemv_last_kernel<<<(N + TB - 1) / TB, TB, 0, stream>>>(h, n_out, W_last, hw3, N);
    last_softmax_pool_kernel<<<(N + TB - 1) / TB, TB, 0, stream>>>(
        hw3, esrc, row_off, n_in_, b_last, pool_w, pool_b, colors, stats, N);
    finalize_kernel<<<1, 1, 0, stream>>>(stats, out_w, out_b, sat);
}
